// _ShareAdaptiveAttention_37211596652544
// MI455X (gfx1250) — matrix operand order repaired and checked by reading, not compiled, not run on hardware
//
#include <hip/hip_runtime.h>
#include <hip/hip_bf16.h>

typedef __bf16 bf16_t;
typedef __attribute__((ext_vector_type(16))) __bf16 v16bf;
typedef __attribute__((ext_vector_type(8)))  __bf16 v8bf;
typedef __attribute__((ext_vector_type(8)))  float  v8f;

// ---------------------------------------------------------------------------
// WMMA helpers (CDNA5 v_wmma_f32_16x16x32_bf16, wave32)
// A fragment (16x32): lane m = lane&15, half = lane>>4.
//   elems 0..7  = K = half*8 + 0..7
//   elems 8..15 = K = 16 + half*8 + 0..7
// B fragment (32x16): lane n = lane&15, same K order as A fragment.
// D (16x16 f32): reg r -> m = half*8 + r, n = lane&15.
// ---------------------------------------------------------------------------
__device__ __forceinline__ v8f wmma_bf16(v16bf a, v16bf b, v8f c) {
  return __builtin_amdgcn_wmma_f32_16x16x32_bf16(
      /*neg_a=*/false, a, /*neg_b=*/false, b,
      /*c_mod=*/(short)0, c, /*reuse_a=*/false, /*reuse_b=*/false);
}

__device__ __forceinline__ v16bf load_a_frag(const bf16_t* __restrict__ src, int ld) {
  const int lane = threadIdx.x & 31;
  const bf16_t* p = src + (size_t)(lane & 15) * ld + ((lane >> 4) * 8);
  v8bf lo = *(const v8bf*)(p);
  v8bf hi = *(const v8bf*)(p + 16);
  v16bf f;
#pragma unroll
  for (int i = 0; i < 8; ++i) { f[i] = lo[i]; f[i + 8] = hi[i]; }
  return f;
}

__device__ __forceinline__ v16bf load_b_frag(const bf16_t* __restrict__ src, int ld) {
  const int lane = threadIdx.x & 31;
  const bf16_t* p = src + (size_t)(lane & 15) * ld + ((lane >> 4) * 8);
  v8bf lo = *(const v8bf*)(p);
  v8bf hi = *(const v8bf*)(p + 16);
  v16bf f;
#pragma unroll
  for (int i = 0; i < 8; ++i) { f[i] = lo[i]; f[i + 8] = hi[i]; }
  return f;
}

// ---------------------------------------------------------------------------
// K0a: f32 -> bf16 bulk convert (weights)
// ---------------------------------------------------------------------------
__global__ void cvt_f32_bf16(const float* __restrict__ in, bf16_t* __restrict__ out, int n) {
  int i = blockIdx.x * blockDim.x + threadIdx.x;
  if (i < n) out[i] = (bf16_t)in[i];
}

// ---------------------------------------------------------------------------
// K0b: base_feat [b][c][l] f32 -> qm_bf [b][l][c] bf16 (LDS tiled transpose)
// grid (Lq/32, C/32, B), block (32, 8)
// ---------------------------------------------------------------------------
__global__ void transpose_to_bf16(const float* __restrict__ base, bf16_t* __restrict__ qm) {
  __shared__ float tile[32][33];
  const int b = blockIdx.z;
  const int l0 = blockIdx.x * 32, c0 = blockIdx.y * 32;
  const int tx = threadIdx.x, ty = threadIdx.y;
  const float* src = base + ((size_t)b * 1024 + c0) * 4096 + l0;
#pragma unroll
  for (int j = 0; j < 32; j += 8)
    tile[ty + j][tx] = src[(size_t)(ty + j) * 4096 + tx];
  __syncthreads();
  bf16_t* dst = qm + ((size_t)b * 4096 + l0) * 1024 + c0;
#pragma unroll
  for (int j = 0; j < 32; j += 8)
    dst[(size_t)(ty + j) * 1024 + tx] = (bf16_t)tile[tx][ty + j];
}

// ---------------------------------------------------------------------------
// K1: support + positional encoding -> s_row [sb][416][1024] and
//     s_colT [sb][1024][416] (both bf16, K-pad rows zeroed)
// support_feat layout: [b][s][c][k], k = 0..399
// ---------------------------------------------------------------------------
__global__ void sprep(const float* __restrict__ sup,
                      bf16_t* __restrict__ s_row, bf16_t* __restrict__ s_colT) {
  int idx = blockIdx.x * blockDim.x + threadIdx.x;   // < 20*416*1024
  int c  = idx & 1023;
  int k  = (idx >> 10) % 416;
  int sb = idx / (416 * 1024);                       // sb = s*4 + b
  int s = sb >> 2, b = sb & 3;
  float v = 0.0f;
  if (k < 400) {
    // pe[k, c]: freq = exp(-(c & ~1) * ln(10000)/1024); even->sin, odd->cos
    float freq = __expf(-(float)(c & ~1) * (9.210340371976184f / 1024.0f));
    float ang = (float)k * freq;
    float pe = (c & 1) ? __cosf(ang) : __sinf(ang);
    v = sup[(((size_t)b * 5 + s) * 1024 + c) * 400 + k] + pe;
  }
  bf16_t bv = (bf16_t)v;
  s_row[((size_t)sb * 416 + k) * 1024 + c] = bv;
  s_colT[((size_t)sb * 1024 + c) * 416 + k] = bv;
}

// ---------------------------------------------------------------------------
// K2: k-projection GEMM: kproj[h,s,b] (416x256) = s_row[sb] (416x1024) @ Kw[h]^T + Kb
// one wave per 32x32 output tile; grid (13*8, 80), block 32
// ---------------------------------------------------------------------------
__global__ void __launch_bounds__(32) kproj_gemm(
    const bf16_t* __restrict__ s_row, const bf16_t* __restrict__ kw_bf,
    const float* __restrict__ kb, bf16_t* __restrict__ kproj) {
  const int tile = blockIdx.x;           // 0..103
  const int mt = tile % 13, nt = tile / 13;
  const int hsb = blockIdx.y;            // 0..79 = h*20 + sb
  const int h = hsb / 20, sb = hsb % 20;
  const bf16_t* A  = s_row + ((size_t)sb * 416 + mt * 32) * 1024;
  const bf16_t* Bm = kw_bf + ((size_t)h * 256 + nt * 32) * 1024;
  v8f acc[2][2] = {};
  for (int k = 0; k < 1024; k += 32) {
    v16bf a0 = load_a_frag(A + k, 1024);
    v16bf a1 = load_a_frag(A + (size_t)16 * 1024 + k, 1024);
    v16bf b0 = load_b_frag(Bm + k, 1024);
    v16bf b1 = load_b_frag(Bm + (size_t)16 * 1024 + k, 1024);
    acc[0][0] = wmma_bf16(a0, b0, acc[0][0]);
    acc[0][1] = wmma_bf16(a0, b1, acc[0][1]);
    acc[1][0] = wmma_bf16(a1, b0, acc[1][0]);
    acc[1][1] = wmma_bf16(a1, b1, acc[1][1]);
  }
  const int lane = threadIdx.x & 31, half = lane >> 4, nn = lane & 15;
  bf16_t* D = kproj + ((size_t)hsb * 416 + mt * 32) * 256 + nt * 32;
#pragma unroll
  for (int mi = 0; mi < 2; ++mi)
#pragma unroll
    for (int ni = 0; ni < 2; ++ni)
#pragma unroll
      for (int r = 0; r < 8; ++r) {
        int m = mi * 16 + half * 8 + r;
        int n = ni * 16 + nn;
        D[(size_t)m * 256 + n] = (bf16_t)(acc[mi][ni][r] + kb[h * 256 + nt * 32 + n]);
      }
}

// ---------------------------------------------------------------------------
// K3: fused attention. One WG per (l-tile of 32, b, h); 8 waves.
//  q-proj (WMMA, into LDS) -> scores (WMMA, into LDS) -> softmax -> bf16 attn
//  -> z GEMM (WMMA, acc over 5 shots) -> mean -> zcat[b*Lq+l][h*1024+c] bf16
// ---------------------------------------------------------------------------
__global__ void __launch_bounds__(256, 1) attn_fused(
    const bf16_t* __restrict__ qm_bf, const bf16_t* __restrict__ qw_bf,
    const float* __restrict__ qb, const bf16_t* __restrict__ kproj,
    const bf16_t* __restrict__ s_colT, bf16_t* __restrict__ zcat) {
  __shared__ __align__(16) bf16_t q_lds[32 * 256];   // 16 KB
  __shared__ __align__(16) float  sc_lds[32 * 416];  // 53.25 KB
  __shared__ __align__(16) bf16_t at_lds[32 * 416];  // 26.6 KB

  const int l0 = blockIdx.x * 32;
  const int b  = blockIdx.y;
  const int h  = blockIdx.z;
  const int w  = threadIdx.x >> 5;
  const int lane = threadIdx.x & 31;
  const int half = lane >> 4, nn = lane & 15;

  // ---- q projection: wave w computes d-range [w*32, w*32+32) of 32x256 tile
  {
    const bf16_t* A  = qm_bf + ((size_t)b * 4096 + l0) * 1024;
    const bf16_t* Bm = qw_bf + ((size_t)h * 256 + w * 32) * 1024;
    v8f acc[2][2] = {};
    for (int k = 0; k < 1024; k += 32) {
      v16bf a0 = load_a_frag(A + k, 1024);
      v16bf a1 = load_a_frag(A + (size_t)16 * 1024 + k, 1024);
      v16bf b0 = load_b_frag(Bm + k, 1024);
      v16bf b1 = load_b_frag(Bm + (size_t)16 * 1024 + k, 1024);
      acc[0][0] = wmma_bf16(a0, b0, acc[0][0]);
      acc[0][1] = wmma_bf16(a0, b1, acc[0][1]);
      acc[1][0] = wmma_bf16(a1, b0, acc[1][0]);
      acc[1][1] = wmma_bf16(a1, b1, acc[1][1]);
    }
#pragma unroll
    for (int mi = 0; mi < 2; ++mi)
#pragma unroll
      for (int ni = 0; ni < 2; ++ni)
#pragma unroll
        for (int r = 0; r < 8; ++r) {
          int m = mi * 16 + half * 8 + r;
          int d = w * 32 + ni * 16 + nn;
          q_lds[m * 256 + d] = (bf16_t)(acc[mi][ni][r] + qb[h * 256 + d]);
        }
  }
  __syncthreads();

  v8f zacc[2][8] = {};
  for (int s = 0; s < 5; ++s) {
    // ---- scores: 13 n-tiles of 32 over Lk, round-robin across waves
    for (int t = w; t < 13; t += 8) {
      const bf16_t* Bk = kproj + ((size_t)(h * 20 + s * 4 + b) * 416 + t * 32) * 256;
      v8f acc[2][2] = {};
      for (int k = 0; k < 256; k += 32) {
        v16bf a0 = load_a_frag(q_lds + k, 256);
        v16bf a1 = load_a_frag(q_lds + 16 * 256 + k, 256);
        v16bf b0 = load_b_frag(Bk + k, 256);
        v16bf b1 = load_b_frag(Bk + (size_t)16 * 256 + k, 256);
        acc[0][0] = wmma_bf16(a0, b0, acc[0][0]);
        acc[0][1] = wmma_bf16(a0, b1, acc[0][1]);
        acc[1][0] = wmma_bf16(a1, b0, acc[1][0]);
        acc[1][1] = wmma_bf16(a1, b1, acc[1][1]);
      }
#pragma unroll
      for (int mi = 0; mi < 2; ++mi)
#pragma unroll
        for (int ni = 0; ni < 2; ++ni)
#pragma unroll
          for (int r = 0; r < 8; ++r) {
            int m = mi * 16 + half * 8 + r;
            int n = t * 32 + ni * 16 + nn;
            sc_lds[m * 416 + n] = acc[mi][ni][r] * 0.0625f;  // 1/sqrt(256)
          }
    }
    __syncthreads();
    // ---- mask K-pad cols 400..415 with -inf
    for (int i = threadIdx.x; i < 512; i += 256)
      sc_lds[(i >> 4) * 416 + 400 + (i & 15)] = -__builtin_inff();
    __syncthreads();
    // ---- row softmax (wave per row, lane-strided, shfl_xor reductions)
    for (int rr = w; rr < 32; rr += 8) {
      float* row = sc_lds + rr * 416;
      float mx = -__builtin_inff();
      for (int j = lane; j < 416; j += 32) mx = fmaxf(mx, row[j]);
#pragma unroll
      for (int o = 16; o; o >>= 1) mx = fmaxf(mx, __shfl_xor(mx, o, 32));
      float sum = 0.0f;
      for (int j = lane; j < 416; j += 32) {
        float e = __expf(row[j] - mx);
        sum += e;
        row[j] = e;
      }
#pragma unroll
      for (int o = 16; o; o >>= 1) sum += __shfl_xor(sum, o, 32);
      float inv = 1.0f / sum;
      for (int j = lane; j < 416; j += 32)
        at_lds[rr * 416 + j] = (bf16_t)(row[j] * inv);
    }
    __syncthreads();
    // ---- z accumulate: wave w covers c-range [w*128, w*128+128)
    const bf16_t* Bv = s_colT + ((size_t)(s * 4 + b) * 1024 + w * 128) * 416;
    for (int kt = 0; kt < 416; kt += 32) {
      v16bf a0 = load_a_frag(at_lds + kt, 416);
      v16bf a1 = load_a_frag(at_lds + 16 * 416 + kt, 416);
#pragma unroll
      for (int ni = 0; ni < 8; ++ni) {
        v16bf bb = load_b_frag(Bv + (size_t)ni * 16 * 416 + kt, 416);
        zacc[0][ni] = wmma_bf16(a0, bb, zacc[0][ni]);
        zacc[1][ni] = wmma_bf16(a1, bb, zacc[1][ni]);
      }
    }
    __syncthreads();
  }
  // ---- mean over shots, write zcat
#pragma unroll
  for (int mi = 0; mi < 2; ++mi)
#pragma unroll
    for (int ni = 0; ni < 8; ++ni)
#pragma unroll
      for (int r = 0; r < 8; ++r) {
        int m = mi * 16 + half * 8 + r;
        int c = w * 128 + ni * 16 + nn;
        zcat[((size_t)b * 4096 + l0 + m) * 4096 + h * 1024 + c] =
            (bf16_t)(zacc[mi][ni][r] * 0.2f);
      }
}

// ---------------------------------------------------------------------------
// K4: att = relu(zcat @ Mw^T + Mb); out[b][co][l] = base[b][co][l] * att
// one wave per 64x64 tile (4x4 WMMA grid, 16 FLOP-dense WMMAs per K-step);
// grid (Lq/64=64, 1024/64=16, B=4), block 32
// ---------------------------------------------------------------------------
__global__ void __launch_bounds__(32) final_gemm(
    const bf16_t* __restrict__ zcat, const bf16_t* __restrict__ mw_bf,
    const float* __restrict__ mb, const float* __restrict__ base,
    float* __restrict__ out) {
  const int l0 = blockIdx.x * 64;
  const int co0 = blockIdx.y * 64;
  const int b = blockIdx.z;
  const bf16_t* A  = zcat + ((size_t)b * 4096 + l0) * 4096;
  const bf16_t* Bm = mw_bf + (size_t)co0 * 4096;
  v8f acc[4][4] = {};
  for (int k = 0; k < 4096; k += 32) {
    v16bf af[4], bfr[4];
#pragma unroll
    for (int i = 0; i < 4; ++i) {
      af[i]  = load_a_frag(A + (size_t)(i * 16) * 4096 + k, 4096);
      bfr[i] = load_b_frag(Bm + (size_t)(i * 16) * 4096 + k, 4096);
    }
#pragma unroll
    for (int mi = 0; mi < 4; ++mi)
#pragma unroll
      for (int ni = 0; ni < 4; ++ni)
        acc[mi][ni] = wmma_bf16(af[mi], bfr[ni], acc[mi][ni]);
  }
  const int lane = threadIdx.x & 31, half = lane >> 4, nn = lane & 15;
#pragma unroll
  for (int mi = 0; mi < 4; ++mi)
#pragma unroll
    for (int ni = 0; ni < 4; ++ni)
#pragma unroll
      for (int r = 0; r < 8; ++r) {
        int l = l0 + mi * 16 + half * 8 + r;
        int co = co0 + ni * 16 + nn;
        float v = acc[mi][ni][r] + mb[co];
        v = v > 0.0f ? v : 0.0f;
        size_t o = ((size_t)b * 1024 + co) * 4096 + l;
        out[o] = base[o] * v;
      }
}

// ---------------------------------------------------------------------------
extern "C" void kernel_launch(void* const* d_in, const int* in_sizes, int n_in,
                              void* d_out, int out_size, void* d_ws, size_t ws_size,
                              hipStream_t stream) {
  (void)in_sizes; (void)n_in; (void)out_size; (void)ws_size;
  const float* base = (const float*)d_in[0];
  const float* sup  = (const float*)d_in[1];
  const float* Qw   = (const float*)d_in[2];
  const float* Qb   = (const float*)d_in[3];
  const float* Kw   = (const float*)d_in[4];
  const float* Kb   = (const float*)d_in[5];
  const float* Mw   = (const float*)d_in[6];
  const float* Mb   = (const float*)d_in[7];
  float* out = (float*)d_out;

  char* ws = (char*)d_ws;
  size_t off = 0;
  auto carve = [&](size_t bytes) -> char* {
    char* p = ws + off;
    off += (bytes + 255) & ~(size_t)255;
    return p;
  };
  bf16_t* qm_bf  = (bf16_t*)carve((size_t)4 * 4096 * 1024 * 2);   // [b][l][c]
  bf16_t* s_row  = (bf16_t*)carve((size_t)20 * 416 * 1024 * 2);   // [sb][k][c]
  bf16_t* s_colT = (bf16_t*)carve((size_t)20 * 1024 * 416 * 2);   // [sb][c][k]
  bf16_t* kproj  = (bf16_t*)carve((size_t)80 * 416 * 256 * 2);    // [hsb][k][d]
  bf16_t* zcat   = (bf16_t*)carve((size_t)4 * 4096 * 4096 * 2);   // [b*Lq+l][h*C+c]
  bf16_t* qw_bf  = (bf16_t*)carve((size_t)4 * 256 * 1024 * 2);
  bf16_t* kw_bf  = (bf16_t*)carve((size_t)4 * 256 * 1024 * 2);
  bf16_t* mw_bf  = (bf16_t*)carve((size_t)1024 * 4096 * 2);

  cvt_f32_bf16<<<(1048576 + 255) / 256, 256, 0, stream>>>(Qw, qw_bf, 1048576);
  cvt_f32_bf16<<<(1048576 + 255) / 256, 256, 0, stream>>>(Kw, kw_bf, 1048576);
  cvt_f32_bf16<<<(4194304 + 255) / 256, 256, 0, stream>>>(Mw, mw_bf, 4194304);
  transpose_to_bf16<<<dim3(128, 32, 4), dim3(32, 8), 0, stream>>>(base, qm_bf);
  sprep<<<(20 * 416 * 1024) / 256, 256, 0, stream>>>(sup, s_row, s_colT);
  kproj_gemm<<<dim3(104, 80), 32, 0, stream>>>(s_row, kw_bf, Kb, kproj);
  attn_fused<<<dim3(128, 4, 4), 256, 0, stream>>>(qm_bf, qw_bf, Qb, kproj, s_colT, zcat);
  final_gemm<<<dim3(64, 16, 4), 32, 0, stream>>>(zcat, mw_bf, Mb, base, out);
}
